// BagNet_79517024518392
// MI455X (gfx1250) — matrix operand order repaired and checked by reading, compiled, not run on hardware
//
#include <hip/hip_runtime.h>
#include <hip/hip_bf16.h>

typedef _Float16 half_t;
typedef __attribute__((ext_vector_type(16))) _Float16 v16h;
typedef __attribute__((ext_vector_type(8)))  _Float16 v8h;
typedef __attribute__((ext_vector_type(8)))  float    v8f;

#define NPATCH 1152
#define KF     61952      // 128*22*22, FC K
#define NF_PAD 112        // 100 padded to 7 N-tiles
#define KSLICE 8          // FC split-K factor (1936/8 = 242 k-steps per slice)

// ---------------------------------------------------------------------------
// BN fold: scale = g*rsqrt(v+eps); shift = be - (m - b)*scale
// layer output = relu(conv(x,w)*scale + shift)   (matches reference algebra)
// ---------------------------------------------------------------------------
__global__ void bn_prep_kernel(const float* __restrict__ b, const float* __restrict__ g,
                               const float* __restrict__ be, const float* __restrict__ m,
                               const float* __restrict__ v,
                               float* __restrict__ scale, float* __restrict__ shift, int n) {
  int i = blockIdx.x * blockDim.x + threadIdx.x;
  if (i < n) {
    float s = g[i] * rsqrtf(v[i] + 1e-5f);
    scale[i] = s;
    shift[i] = be[i] - (m[i] - b[i]) * s;
  }
}

// (cout,cin,3,3) f32  ->  [cout][kyx][cin] f16   (B-operand friendly: K=(kyx,ci))
__global__ void conv_w_reorder(const float* __restrict__ w, half_t* __restrict__ wr,
                               int cout, int cin) {
  int i = blockIdx.x * blockDim.x + threadIdx.x;
  int total = cout * cin * 9;
  if (i >= total) return;
  int c   = i / (cin * 9);
  int r   = i % (cin * 9);
  int ci  = r / 9;
  int kyx = r % 9;
  wr[(size_t)c * cin * 9 + (size_t)kyx * cin + ci] = (half_t)w[i];
}

// wf (100, k_old = c*484 + yx)  ->  wfr [112][k_new = yx*128 + c], zero-padded rows
__global__ void fc_w_reorder(const float* __restrict__ wf, half_t* __restrict__ wfr) {
  int i = blockIdx.x * blockDim.x + threadIdx.x;         // over 112*61952
  if (i >= NF_PAD * KF) return;
  int o = i / KF;
  int k = i % KF;
  int c  = k & 127;
  int yx = k >> 7;
  half_t val = (half_t)0.f;
  if (o < 100) val = (half_t)wf[(size_t)o * KF + (size_t)c * 484 + yx];
  wfr[i] = val;
}

// ---------------------------------------------------------------------------
// Layer 1: patchify + 3x3 conv (Cin=1, K=9 -> VALU) + BN + ReLU, h1 NHWC f16
// ---------------------------------------------------------------------------
__global__ void conv1_kernel(const float* __restrict__ x, const float* __restrict__ w1,
                             const float* __restrict__ scale1, const float* __restrict__ shift1,
                             half_t* __restrict__ h1) {
  __shared__ float ws[64 * 9];
  __shared__ float sc[64], sh[64];
  for (int i = threadIdx.x; i < 64 * 9; i += blockDim.x) ws[i] = w1[i];
  for (int i = threadIdx.x; i < 64; i += blockDim.x) { sc[i] = scale1[i]; sh[i] = shift1[i]; }
  __syncthreads();

  int gid = blockIdx.x * blockDim.x + threadIdx.x;       // over 1152*676
  if (gid >= NPATCH * 676) return;
  int patch = gid / 676;
  int p = gid % 676;
  int oy = p / 26, ox = p % 26;
  int b  = patch / 9;
  int hb = (patch % 9) / 3;
  int wb = patch % 3;
  const float* xb = x + (size_t)b * 84 * 84 + (size_t)(hb * 28 + oy) * 84 + (wb * 28 + ox);

  float in[9];
#pragma unroll
  for (int ky = 0; ky < 3; ky++)
#pragma unroll
    for (int kx = 0; kx < 3; kx++) in[ky * 3 + kx] = xb[ky * 84 + kx];

  half_t* out = h1 + (size_t)gid * 64;                   // [patch][yx][c]
#pragma unroll 4
  for (int c = 0; c < 64; c++) {
    float acc = 0.f;
#pragma unroll
    for (int t = 0; t < 9; t++) acc += in[t] * ws[c * 9 + t];
    float r = acc * sc[c] + sh[c];
    out[c] = (half_t)(r > 0.f ? r : 0.f);
  }
}

// ---------------------------------------------------------------------------
// Implicit-GEMM conv (Cin=64) via v_wmma_f32_16x16x32_f16.
// One wave = 16(M=spatial) x 64(N=cout) tile: one A fragment feeds 4 WMMAs.
// All geometry is compile-time -> no integer-division decode.
// in:  [patch][Hin][Hin][64] f16
// wr:  [cout][kyx][64] f16
// out: [patch][Hout^2][Cout] f16
// ---------------------------------------------------------------------------
template <int Hin, int Hout, int Cout, int Mtiles, int NGroups>
__global__ void conv_wmma_kernel(const half_t* __restrict__ in, const half_t* __restrict__ wr,
                                 const float* __restrict__ scale, const float* __restrict__ shift,
                                 half_t* __restrict__ out) {
  constexpr int Mpos = Hout * Hout;
  constexpr int TPP  = Mtiles * NGroups;       // wave-tiles per patch

  int lane = threadIdx.x & 31;
  int wave = threadIdx.x >> 5;
  int gw = blockIdx.x * (blockDim.x >> 5) + wave;
  int patch = gw / TPP;                        // division by constant -> mul/shift
  if (patch >= NPATCH) return;
  int t  = gw % TPP;
  int tm = t / NGroups;
  int ng = t % NGroups;

  int half_id = lane >> 4;
  int lid     = lane & 15;

  // A row (spatial position) for this lane
  int m  = tm * 16 + lid;
  int mc = (m < Mpos) ? m : 0;                 // clamp ragged last M tile (store guarded)
  int oy = mc / Hout, ox = mc % Hout;
  const half_t* inP = in + (size_t)patch * Hin * Hin * 64;

  // 4 B columns (output channels) for this lane
  const half_t* wP[4];
  float sc[4], sh[4];
#pragma unroll
  for (int n = 0; n < 4; n++) {
    int cN = ng * 64 + n * 16 + lid;           // NGroups*64 == Cout, always in range
    wP[n] = wr + (size_t)cN * 576;
    sc[n] = scale[cN];
    sh[n] = shift[cN];
  }

  v8f acc[4] = {};
#pragma unroll
  for (int s = 0; s < 18; s++) {
    constexpr int dummy = 0; (void)dummy;
    int kyx = s >> 1;
    int ky = kyx / 3, kx = kyx - ky * 3;       // folds at compile time (full unroll)
    int cib = (s & 1) * 32;

    // A fragment: k-locals {8h..8h+7} and {16+8h..23+8h} -> two 16B contiguous loads
    const half_t* ap = inP + ((size_t)(oy + ky) * Hin + (ox + kx)) * 64 + cib;
    union { v16h v; v8h h[2]; } A;
    A.h[0] = *(const v8h*)(ap + 8 * half_id);
    A.h[1] = *(const v8h*)(ap + 16 + 8 * half_id);

#pragma unroll
    for (int n = 0; n < 4; n++) {
      const half_t* bp = wP[n] + kyx * 64 + cib + 8 * half_id;
      union { v16h v; v8h h[2]; } B;
      B.h[0] = *(const v8h*)(bp);
      B.h[1] = *(const v8h*)(bp + 16);
      acc[n] = __builtin_amdgcn_wmma_f32_16x16x32_f16(false, A.v, false, B.v,
                                                      (short)0, acc[n], false, false);
    }
  }

  // Epilogue: D vgpr i -> row M = tm*16 + 8*half_id + i, col N = cN ; BN+ReLU, f16 NHWC
  half_t* outP = out + (size_t)patch * Mpos * Cout;
#pragma unroll
  for (int i = 0; i < 8; i++) {
    int mm = tm * 16 + half_id * 8 + i;
    if (mm < Mpos) {
      half_t* row = outP + (size_t)mm * Cout + ng * 64 + lid;
#pragma unroll
      for (int n = 0; n < 4; n++) {
        float r = acc[n][i] * sc[n] + sh[n];
        row[n * 16] = (half_t)(r > 0.f ? r : 0.f);
      }
    }
  }
}

// ---------------------------------------------------------------------------
// FC split-K: partial[slice][patch][o] = h3[patch, kslice] @ wfr[o, kslice]^T
// 72 M-tiles x 7 N-tiles x 8 K-slices = 4032 waves; deterministic (fixed slices)
// ---------------------------------------------------------------------------
__global__ void fc_wmma_kernel(const half_t* __restrict__ h3, const half_t* __restrict__ wfr,
                               float* __restrict__ partials) {
  constexpr int STEPS = (KF / 32) / KSLICE;    // 242 k-steps per slice
  int lane = threadIdx.x & 31;
  int wave = threadIdx.x >> 5;
  int gw = blockIdx.x * (blockDim.x >> 5) + wave;
  if (gw >= 72 * 7 * KSLICE) return;
  int tm    = gw / (7 * KSLICE);
  int rem   = gw % (7 * KSLICE);
  int tn    = rem / KSLICE;
  int slice = rem % KSLICE;

  int half_id = lane >> 4;
  int lid     = lane & 15;

  int mP = tm * 16 + lid;                      // patch (A row), always < 1152
  const half_t* aP = h3 + (size_t)mP * KF + (size_t)slice * STEPS * 32;
  int o = tn * 16 + lid;                       // output neuron (B col / D col)
  const half_t* bP = wfr + (size_t)o * KF + (size_t)slice * STEPS * 32;

  v8f acc = {};
#pragma unroll 2
  for (int s = 0; s < STEPS; s++) {
    const half_t* ap = aP + s * 32;
    union { v16h v; v8h h[2]; } A;
    A.h[0] = *(const v8h*)(ap + 8 * half_id);
    A.h[1] = *(const v8h*)(ap + 16 + 8 * half_id);
    const half_t* bp = bP + s * 32 + 8 * half_id;
    union { v16h v; v8h h[2]; } B;
    B.h[0] = *(const v8h*)(bp);
    B.h[1] = *(const v8h*)(bp + 16);
    acc = __builtin_amdgcn_wmma_f32_16x16x32_f16(false, A.v, false, B.v,
                                                 (short)0, acc, false, false);
  }

  float* pP = partials + (size_t)slice * NPATCH * NF_PAD;
#pragma unroll
  for (int i = 0; i < 8; i++) {
    int mm = tm * 16 + half_id * 8 + i;        // patch row
    pP[(size_t)mm * NF_PAD + o] = acc[i];
  }
}

// Fixed-order reduction over the 8 K-slices + bias (bit-deterministic).
__global__ void fc_reduce_kernel(const float* __restrict__ partials,
                                 const float* __restrict__ bf, float* __restrict__ outp) {
  int i = blockIdx.x * blockDim.x + threadIdx.x;         // over 1152*100
  if (i >= NPATCH * 100) return;
  int p = i / 100;
  int o = i % 100;
  float acc = bf[o];
#pragma unroll
  for (int s = 0; s < KSLICE; s++)
    acc += partials[(size_t)s * NPATCH * NF_PAD + (size_t)p * NF_PAD + o];
  outp[i] = acc;
}

// ---------------------------------------------------------------------------
extern "C" void kernel_launch(void* const* d_in, const int* in_sizes, int n_in,
                              void* d_out, int out_size, void* d_ws, size_t ws_size,
                              hipStream_t stream) {
  const float* x   = (const float*)d_in[0];
  const float* w1  = (const float*)d_in[1];
  const float* b1  = (const float*)d_in[2];
  const float* g1  = (const float*)d_in[3];
  const float* be1 = (const float*)d_in[4];
  const float* m1  = (const float*)d_in[5];
  const float* v1  = (const float*)d_in[6];
  const float* w2  = (const float*)d_in[7];
  const float* b2  = (const float*)d_in[8];
  const float* g2  = (const float*)d_in[9];
  const float* be2 = (const float*)d_in[10];
  const float* m2  = (const float*)d_in[11];
  const float* v2  = (const float*)d_in[12];
  const float* w3  = (const float*)d_in[13];
  const float* b3  = (const float*)d_in[14];
  const float* g3  = (const float*)d_in[15];
  const float* be3 = (const float*)d_in[16];
  const float* m3  = (const float*)d_in[17];
  const float* v3  = (const float*)d_in[18];
  const float* wf  = (const float*)d_in[19];
  const float* bf  = (const float*)d_in[20];

  // workspace carve-up (256B aligned; total ~346 MB)
  char* ws = (char*)d_ws;
  size_t off = 0;
  auto carve = [&](size_t bytes) -> char* {
    char* p = ws + off;
    off = (off + bytes + 255) & ~(size_t)255;
    return p;
  };
  float*  scale1 = (float*)carve(64 * 4);
  float*  shift1 = (float*)carve(64 * 4);
  float*  scale2 = (float*)carve(64 * 4);
  float*  shift2 = (float*)carve(64 * 4);
  float*  scale3 = (float*)carve(128 * 4);
  float*  shift3 = (float*)carve(128 * 4);
  half_t* w2r = (half_t*)carve((size_t)64 * 576 * 2);             //  72 KB
  half_t* w3r = (half_t*)carve((size_t)128 * 576 * 2);            // 144 KB
  half_t* wfr = (half_t*)carve((size_t)NF_PAD * KF * 2);          // 13.9 MB
  half_t* h1  = (half_t*)carve((size_t)NPATCH * 676 * 64 * 2);    // 99.7 MB
  half_t* h2  = (half_t*)carve((size_t)NPATCH * 576 * 64 * 2);    // 84.9 MB
  half_t* h3  = (half_t*)carve((size_t)NPATCH * 484 * 128 * 2);   // 142.7 MB
  float*  fcp = (float*)carve((size_t)KSLICE * NPATCH * NF_PAD * 4); // 4.1 MB

  // BN folds
  bn_prep_kernel<<<1, 64,  0, stream>>>(b1, g1, be1, m1, v1, scale1, shift1, 64);
  bn_prep_kernel<<<1, 64,  0, stream>>>(b2, g2, be2, m2, v2, scale2, shift2, 64);
  bn_prep_kernel<<<1, 128, 0, stream>>>(b3, g3, be3, m3, v3, scale3, shift3, 128);

  // weight reorders to f16 WMMA-friendly layouts
  conv_w_reorder<<<(64 * 64 * 9 + 255) / 256, 256, 0, stream>>>(w2, w2r, 64, 64);
  conv_w_reorder<<<(128 * 64 * 9 + 255) / 256, 256, 0, stream>>>(w3, w3r, 128, 64);
  fc_w_reorder<<<(NF_PAD * KF) / 256, 256, 0, stream>>>(wf, wfr);

  // layer 1 (VALU, Cin=1)
  conv1_kernel<<<(NPATCH * 676 + 255) / 256, 256, 0, stream>>>(x, w1, scale1, shift1, h1);

  // layer 2: 1152 * 36 M-tiles * 1 N-group (16x64 per wave) = 41472 waves
  conv_wmma_kernel<26, 24, 64, 36, 1><<<NPATCH * 36 * 1 / 8, 256, 0, stream>>>(
      h1, w2r, scale2, shift2, h2);

  // layer 3: 1152 * 31 M-tiles * 2 N-groups = 71424 waves
  conv_wmma_kernel<24, 22, 128, 31, 2><<<NPATCH * 31 * 2 / 8, 256, 0, stream>>>(
      h2, w3r, scale3, shift3, h3);

  // FC: split-K partials then fixed-order reduce
  fc_wmma_kernel<<<72 * 7 * KSLICE / 8, 256, 0, stream>>>(h3, wfr, fcp);
  fc_reduce_kernel<<<(NPATCH * 100 + 255) / 256, 256, 0, stream>>>(fcp, bf, (float*)d_out);
}
